// kNNSpatialConvolution_41747082117333
// MI455X (gfx1250) — matrix operand order repaired and checked by reading, not compiled, not run on hardware
//
#include <hip/hip_runtime.h>
#include <math.h>

typedef _Float16 half_t;
typedef __attribute__((ext_vector_type(16))) _Float16 v16h;
typedef __attribute__((ext_vector_type(8)))  float    v8f;

// ---------------- problem constants ----------------
#define NPTS 8192
#define KNN  16
#define M0   128
#define M1   64
#define RB   42
#define RCUT 32.0f

// padded contraction dims (multiples of 32 for 16x16x32 WMMA)
#define C0P 224   // in_s cols: 193 -> 224
#define C1P 288   // in_v rows: 257 -> 288
#define CMP 320   // mix_in cols: 298 -> 320
#define HP  64    // hidden dim: 42 -> 64

// workspace layout (halves for transposed padded f16 weights, then ints/floats)
#define W0T_H (128*224)
#define W1T_H (64*288)
#define M1T_H (48*320)
#define M2T_H (192*64)
#define WTOT_H (W0T_H + W1T_H + M1T_H + M2T_H)          // 74752 halves = 149504 B
#define NEI_OFF_B   149504u
#define NMASK_OFF_B (NEI_OFF_B + (size_t)NPTS*KNN*4)

#define BIGF 3.0e38f

// ---------------- WMMA tile loaders ----------------
// A (16x32 f16): lane L -> row m=L&15, group g=L>>4; element e -> col 16*(e>>3)+8*g+(e&7)
__device__ __forceinline__ v16h loadA16(const half_t* Abase, int stride, int kc, int lane) {
  const int g = lane >> 4, m = lane & 15;
  const half_t* p = Abase + m * stride + kc * 32 + 8 * g;
  union { v16h v; uint4 q[2]; } u;
  u.q[0] = *(const uint4*)(p);
  u.q[1] = *(const uint4*)(p + 16);
  return u.v;
}
// B (32x16 f16) from column-major padded weight Wt[col][rowPad]:
// lane L -> col n=L&15 (+tile), group g=L>>4; element e -> row kc*32+16*(e>>3)+8*g+(e&7)
__device__ __forceinline__ v16h loadB16(const half_t* Wt, int rowsPad, int kc, int ntile, int lane) {
  const int g = lane >> 4, nn = lane & 15;
  const half_t* p = Wt + (size_t)(ntile * 16 + nn) * rowsPad + kc * 32 + 8 * g;
  union { v16h v; uint4 q[2]; } u;
  u.q[0] = *(const uint4*)(p);
  u.q[1] = *(const uint4*)(p + 16);
  return u.v;
}

// ---------------- kernel 0: convert/transpose/pad weights to f16 ----------------
__global__ void prep_weights(const float* __restrict__ W0, const float* __restrict__ W1,
                             const float* __restrict__ Mw1, const float* __restrict__ Mw2,
                             half_t* __restrict__ wsH) {
  int i = blockIdx.x * blockDim.x + threadIdx.x;
  if (i < W0T_H) {                                   // W0t[128][224]
    int c = i / 224, r = i % 224;
    wsH[i] = (r < 193) ? (half_t)W0[r * 128 + c] : (half_t)0.f;
    return;
  }
  i -= W0T_H;
  if (i < W1T_H) {                                   // W1t[64][288]
    int c = i / 288, r = i % 288;
    wsH[W0T_H + i] = (r < 257) ? (half_t)W1[r * 64 + c] : (half_t)0.f;
    return;
  }
  i -= W1T_H;
  if (i < M1T_H) {                                   // M1t[48][320]
    int c = i / 320, r = i % 320;
    wsH[W0T_H + W1T_H + i] = (c < 42 && r < 298) ? (half_t)Mw1[r * 42 + c] : (half_t)0.f;
    return;
  }
  i -= M1T_H;
  if (i < M2T_H) {                                   // M2t[192][64]
    int c = i / 64, r = i % 64;
    wsH[W0T_H + W1T_H + M1T_H + i] = (r < 42) ? (half_t)Mw2[r * 192 + c] : (half_t)0.f;
  }
}

// ---------------- kernel 1: kNN (one wave32 per node, register-resident top-17) --------
__global__ void knn_kernel(const float* __restrict__ coord, const unsigned char* __restrict__ mask,
                           int* __restrict__ nei, float* __restrict__ nmaskf) {
  const int wid  = (blockIdx.x * blockDim.x + threadIdx.x) >> 5;
  const int lane = threadIdx.x & 31;
  if (wid >= NPTS) return;
  const int n = wid;
  const float cx = coord[3 * n], cy = coord[3 * n + 1], cz = coord[3 * n + 2];
  const bool mn = mask[n] != 0;

  float dl[17]; int il[17];
#pragma unroll
  for (int p = 0; p < 17; ++p) { dl[p] = BIGF; il[p] = 0x7fffffff; }

  for (int j = lane; j < NPTS; j += 32) {
    float d2;
    const bool mj = mask[j] != 0;
    if (!mn || !mj || j == n) {
      d2 = BIGF;
    } else {
      float dx = coord[3 * j] - cx, dy = coord[3 * j + 1] - cy, dz = coord[3 * j + 2] - cz;
      d2 = dx * dx + dy * dy + dz * dz;
      int dd = (j > n) ? (j - n) : (n - j);
      if (dd <= 8) d2 = -BIGF;              // sequence band forced first (dd>=1 since j!=n)
    }
    if (d2 < dl[16] || (d2 == dl[16] && j < il[16])) {
      float vI = d2; int iI = j;
#pragma unroll
      for (int p = 0; p < 17; ++p) {        // unrolled sorted insert -> stays in VGPRs
        bool sw = (vI < dl[p]) || (vI == dl[p] && iI < il[p]);
        if (sw) { float tv = dl[p]; dl[p] = vI; vI = tv; int ti = il[p]; il[p] = iI; iI = ti; }
      }
    }
  }

  // global 17-way selection across the wave (lexicographic min-reduce on (d2, idx))
  for (int s = 0; s < 17; ++s) {
    float v = dl[0]; int jj = il[0];
#pragma unroll
    for (int off = 16; off >= 1; off >>= 1) {
      float ov = __shfl_xor(v, off, 32);
      int   oj = __shfl_xor(jj, off, 32);
      if (ov < v || (ov == v && oj < jj)) { v = ov; jj = oj; }
    }
    if (dl[0] == v && il[0] == jj) {        // owner pops front (unique except sentinels)
#pragma unroll
      for (int p = 0; p < 16; ++p) { dl[p] = dl[p + 1]; il[p] = il[p + 1]; }
      dl[16] = BIGF; il[16] = 0x7fffffff;
    }
    if (s >= 1 && lane == 0) {              // drop first (self/band head), keep 16
      int jo = (jj == 0x7fffffff) ? 0 : jj;
      nei[n * KNN + s - 1]    = jo;
      nmaskf[n * KNN + s - 1] = (v < 1.0e37f) ? 1.f : 0.f;
    }
  }
}

// ---------------- kernel 2: fused message passing (1 block = 1 node, 8 waves) ----------
__global__ __launch_bounds__(256) void fused_kernel(
    const float* __restrict__ coord, const float* __restrict__ feat_s,
    const float* __restrict__ feat_v, const float* __restrict__ mlp_b1,
    const float* __restrict__ mlp_b2, const float* __restrict__ w_move,
    const half_t* __restrict__ wsH, const int* __restrict__ nei,
    const float* __restrict__ nmaskf, float* __restrict__ out) {
  __shared__ half_t sA[16 * CMP];           // in_s (16x224) then mix_in (16x320)   10240 B
  __shared__ half_t sB[3 * 16 * C1P];       // in_v (3x16x288) then mix f32 overlay 27648 B
  __shared__ float  sMsgS[16 * 128];        //                                       8192 B
  __shared__ float  sMsgV[3 * 16 * 64];     //                                      12288 B
  __shared__ half_t sRad[16 * 48];          //                                       1536 B
  __shared__ half_t sH[16 * HP];            //                                       2048 B
  __shared__ float  sOutV[192];             //                                        768 B
  float* sMix = (float*)sB;                 // 16x192 f32 fits in sB region

  const int n = blockIdx.x;
  const int t = threadIdx.x;
  const int wv = t >> 5, lane = t & 31;
  const half_t* W0t = wsH;
  const half_t* W1t = wsH + W0T_H;
  const half_t* M1t = W1t + W1T_H;
  const half_t* M2t = M1t + M1T_H;

  // ---- phase 1: geometry + build in_s / in_v / rad ----
  {
    const int k = t >> 4, h = t & 15;
    int j = nei[n * KNN + k];
    if (j < 0) j = 0; if (j >= NPTS) j = NPTS - 1;
    const float w  = nmaskf[n * KNN + k];
    const float cx = coord[3 * n], cy = coord[3 * n + 1], cz = coord[3 * n + 2];
    const float vx = coord[3 * j] - cx, vy = coord[3 * j + 1] - cy, vz = coord[3 * j + 2] - cz;
    const float n2 = vx * vx + vy * vy + vz * vz;
    const float nr = sqrtf(n2 == 0.f ? 1.f : n2);
    const float shx = w * 1.7320508f * vx, shy = w * 1.7320508f * vy, shz = w * 1.7320508f * vz;
    const float x = nr * (1.f / RCUT);
    for (int b = h; b < 48; b += 16) {
      float val = 0.f;
      if (b < 42 && x > 0.f && x < 1.f) val = sinf(3.14159265f * (float)(b + 1) * x) * 0.2169305f;
      sRad[k * 48 + b] = (half_t)(val * w);
    }
    half_t* inS  = sA + k * C0P;
    half_t* inV0 = sB + 0 * 16 * C1P + k * C1P;
    half_t* inV1 = sB + 1 * 16 * C1P + k * C1P;
    half_t* inV2 = sB + 2 * 16 * C1P + k * C1P;
    for (int c = h; c < 128; c += 16) {
      float fs = w * feat_s[j * M0 + c];               // ns
      inS[c]       = (half_t)(w * fs);                 // tp_s_a
      inV0[64 + c] = (half_t)(shx * fs);               // tp_v_b
      inV1[64 + c] = (half_t)(shy * fs);
      inV2[64 + c] = (half_t)(shz * fs);
    }
    for (int d = h; d < 64; d += 16) {
      const float* fv = feat_v + (size_t)(j * M1 + d) * 3;
      float bx = w * fv[0], by = w * fv[1], bz = w * fv[2];   // nv
      inV0[d] = (half_t)bx; inV1[d] = (half_t)by; inV2[d] = (half_t)bz;     // tp_v_a
      inS[128 + d] = (half_t)((shx * bx + shy * by + shz * bz) * 0.57735027f); // tp_s_b
      inV0[192 + d] = (half_t)((shy * bz - shz * by) * 0.70710678f);        // tp_v_c
      inV1[192 + d] = (half_t)((shz * bx - shx * bz) * 0.70710678f);
      inV2[192 + d] = (half_t)((shx * by - shy * bx) * 0.70710678f);
    }
    if (h == 0) {
      inS[192] = (half_t)w;                            // sh0
      inV0[256] = (half_t)shx; inV1[256] = (half_t)shy; inV2[256] = (half_t)shz;
    }
    for (int c = 193 + h; c < C0P; c += 16) inS[c] = (half_t)0.f;
    for (int c = 257 + h; c < C1P; c += 16) { inV0[c] = (half_t)0.f; inV1[c] = (half_t)0.f; inV2[c] = (half_t)0.f; }
  }
  __syncthreads();

  // ---- phase 2: msg_s = in_s @ W0 / sqrt(193) : 8 tiles, one per wave ----
  {
    v8f acc = {0.f, 0.f, 0.f, 0.f, 0.f, 0.f, 0.f, 0.f};
    for (int kc = 0; kc < C0P / 32; ++kc) {
      v16h a = loadA16(sA, C0P, kc, lane);
      v16h b = loadB16(W0t, C0P, kc, wv, lane);
      acc = __builtin_amdgcn_wmma_f32_16x16x32_f16(false, a, false, b, (short)0, acc, false, false);
    }
    const int g = lane >> 4, nn = lane & 15;
#pragma unroll
    for (int r = 0; r < 8; ++r) sMsgS[(r + 8 * g) * 128 + wv * 16 + nn] = acc[r] * 0.0719816f;
  }
  // ---- phase 3: msg_v = in_v @ W1 / sqrt(257) : 12 tiles (3 comps x 4) ----
  for (int tile = wv; tile < 12; tile += 8) {
    const int comp = tile >> 2, nt = tile & 3;
    v8f acc = {0.f, 0.f, 0.f, 0.f, 0.f, 0.f, 0.f, 0.f};
    for (int kc = 0; kc < C1P / 32; ++kc) {
      v16h a = loadA16(sB + comp * 16 * C1P, C1P, kc, lane);
      v16h b = loadB16(W1t, C1P, kc, nt, lane);
      acc = __builtin_amdgcn_wmma_f32_16x16x32_f16(false, a, false, b, (short)0, acc, false, false);
    }
    const int g = lane >> 4, nn = lane & 15;
#pragma unroll
    for (int r = 0; r < 8; ++r) sMsgV[comp * 1024 + (r + 8 * g) * 64 + nt * 16 + nn] = acc[r] * 0.0623783f;
  }
  __syncthreads();

  // ---- phase 4: build mix_in = [msg_s | rad | feat_s(center)] into sA; zero sH ----
  {
    const int k = t >> 4, h = t & 15;
    half_t* mi = sA + k * CMP;
    for (int c = h; c < 128; c += 16) mi[c] = (half_t)sMsgS[k * 128 + c];
    for (int b = h; b < 42; b += 16) mi[128 + b] = sRad[k * 48 + b];
    for (int c = h; c < 128; c += 16) mi[170 + c] = (half_t)feat_s[n * M0 + c];
    for (int c = 298 + h; c < CMP; c += 16) mi[c] = (half_t)0.f;
#pragma unroll
    for (int q = 0; q < 4; ++q) sH[t * 4 + q] = (half_t)0.f;
  }
  __syncthreads();

  // ---- phase 5: h = silu(mix_in @ mlp_w1 / sqrt(298) + b1) : 3 tiles ----
  if (wv < 3) {
    v8f acc = {0.f, 0.f, 0.f, 0.f, 0.f, 0.f, 0.f, 0.f};
    for (int kc = 0; kc < CMP / 32; ++kc) {
      v16h a = loadA16(sA, CMP, kc, lane);
      v16h b = loadB16(M1t, CMP, kc, wv, lane);
      acc = __builtin_amdgcn_wmma_f32_16x16x32_f16(false, a, false, b, (short)0, acc, false, false);
    }
    const int g = lane >> 4, nn = lane & 15;
    const int col = wv * 16 + nn;
    const float bias = (col < 42) ? mlp_b1[col] : 0.f;
#pragma unroll
    for (int r = 0; r < 8; ++r) {
      float vvl = acc[r] * 0.0579284f + bias;
      float sl = vvl / (1.f + __expf(-vvl));
      sH[(r + 8 * g) * HP + col] = (col < 42) ? (half_t)sl : (half_t)0.f;
    }
  }
  __syncthreads();

  // ---- phase 6: mix = h @ mlp_w2 / sqrt(42) + b2 : 12 tiles (overlay sB as f32) ----
  for (int tile = wv; tile < 12; tile += 8) {
    v8f acc = {0.f, 0.f, 0.f, 0.f, 0.f, 0.f, 0.f, 0.f};
    for (int kc = 0; kc < 2; ++kc) {
      v16h a = loadA16(sH, HP, kc, lane);
      v16h b = loadB16(M2t, HP, kc, tile, lane);
      acc = __builtin_amdgcn_wmma_f32_16x16x32_f16(false, a, false, b, (short)0, acc, false, false);
    }
    const int g = lane >> 4, nn = lane & 15;
    const int col = tile * 16 + nn;
    const float bias = mlp_b2[col];
#pragma unroll
    for (int r = 0; r < 8; ++r) sMix[(r + 8 * g) * 192 + col] = acc[r] * 0.1543033f + bias;
  }
  __syncthreads();

  // ---- phase 7: gated reductions over K and outputs ----
  if (t < 128) {
    float s = 0.f;
#pragma unroll
    for (int k = 0; k < 16; ++k) s += sMsgS[k * 128 + t] * sMix[k * 192 + t];
    out[(size_t)n * M0 + t] = feat_s[(size_t)n * M0 + t] + s * (1.f / 16.f);
  }
  if (t < 192) {
    const int d = t / 3, i = t - 3 * d;
    float s = 0.f;
#pragma unroll
    for (int k = 0; k < 16; ++k) s += sMsgV[i * 1024 + k * 64 + d] * sMix[k * 192 + 128 + d];
    s *= (1.f / 16.f);
    sOutV[t] = s;
    out[(size_t)NPTS * M0 + (size_t)n * 192 + d * 3 + i] =
        feat_v[((size_t)n * M1 + d) * 3 + i] + s;
  }
  __syncthreads();
  if (t < 3) {
    float acc = 0.f;
    for (int d = 0; d < 64; ++d) acc += sOutV[d * 3 + t] * w_move[d];
    out[(size_t)NPTS * (M0 + 192) + (size_t)n * 3 + t] =
        coord[3 * n + t] + 0.001f * acc * (1.f / 8.f);
  }
}

// ---------------- launch ----------------
extern "C" void kernel_launch(void* const* d_in, const int* in_sizes, int n_in,
                              void* d_out, int out_size, void* d_ws, size_t ws_size,
                              hipStream_t stream) {
  (void)in_sizes; (void)n_in; (void)out_size; (void)ws_size;
  const float* coord  = (const float*)d_in[0];
  const float* feat_s = (const float*)d_in[1];
  const float* feat_v = (const float*)d_in[2];
  const unsigned char* mask = (const unsigned char*)d_in[3];
  const float* W0   = (const float*)d_in[4];
  const float* W1   = (const float*)d_in[5];
  const float* mw1  = (const float*)d_in[6];
  const float* mb1  = (const float*)d_in[7];
  const float* mw2  = (const float*)d_in[8];
  const float* mb2  = (const float*)d_in[9];
  const float* wmv  = (const float*)d_in[10];

  half_t* wsH   = (half_t*)d_ws;
  int*    nei   = (int*)((char*)d_ws + NEI_OFF_B);
  float*  nmask = (float*)((char*)d_ws + NMASK_OFF_B);
  float*  out   = (float*)d_out;

  prep_weights<<<(WTOT_H + 255) / 256, 256, 0, stream>>>(W0, W1, mw1, mw2, wsH);
  knn_kernel<<<(NPTS * 32 + 255) / 256, 256, 0, stream>>>(coord, mask, nei, nmask);
  fused_kernel<<<NPTS, 256, 0, stream>>>(coord, feat_s, feat_v, mb1, mb2, wmv,
                                         wsH, nei, nmask, out);
}
